// MultiHeadAttention_14645838479587
// MI455X (gfx1250) — matrix operand order repaired and checked by reading, not compiled, not run on hardware
//
#include <hip/hip_runtime.h>
#include <math.h>

#define D_MODEL 1024
#define NHEADS  16
#define DK      64
#define BATCH   4
#define SEQ     2048
#define M_TOTAL (BATCH*SEQ)   // 8192
#define LDSTR   72            // attention LDS row stride (halves), 144B = 16B-aligned
#define ASTR    56            // gemm LDS row stride (halves), 112B = 16B-aligned
#define KSTEP   32

typedef __attribute__((ext_vector_type(16))) _Float16 v16h;
typedef __attribute__((ext_vector_type(8)))  _Float16 v8h;
typedef __attribute__((ext_vector_type(8)))  float    v8f;
typedef __attribute__((ext_vector_type(4)))  int      v4i;

#if __has_builtin(__builtin_amdgcn_global_load_async_to_lds_b128)
#define HAVE_ASYNC 1
#endif

#define AS1 __attribute__((address_space(1)))
#define AS3 __attribute__((address_space(3)))

// ---- async global->LDS 16B copy (ASYNCcnt path); sync fallback ----
__device__ __forceinline__ void async_ld_b128(const _Float16* g, _Float16* l) {
#ifdef HAVE_ASYNC
  __builtin_amdgcn_global_load_async_to_lds_b128((AS1 v4i*)g, (AS3 v4i*)l, 0, 0);
#else
  *(v8h*)l = *(const v8h*)g;
#endif
}

__device__ __forceinline__ void wait_async0() {
#ifdef HAVE_ASYNC
#if __has_builtin(__builtin_amdgcn_s_wait_asynccnt)
  __builtin_amdgcn_s_wait_asynccnt(0);
#else
  asm volatile("s_wait_asynccnt 0" ::: "memory");
#endif
#endif
}

// ---------------- f32 -> f16 cast ----------------
__global__ void cast_f16_kernel(const float* __restrict__ src,
                                _Float16* __restrict__ dst, int n) {
  int i = blockIdx.x * blockDim.x + threadIdx.x;
  int stride = gridDim.x * blockDim.x;
  for (; i < n; i += stride) dst[i] = (_Float16)src[i];
}

// ---------------- WMMA fragment loaders ----------------
// A 16x32 f16 layout: lane<16 -> M=lane, K chunks [k0,k0+8) and [k0+16,k0+24)
//                     lane>=16 -> M=lane-16, chunks shifted by +8
__device__ __forceinline__ v16h load_a_frag(const _Float16* __restrict__ base,
                                            int ld, int row0, int k0, int lane) {
  int r = lane & 15;
  int sel = lane >> 4;
  const _Float16* p = base + (size_t)(row0 + r) * ld + k0 + sel * 8;
  v8h lo = *(const v8h*)p;
  v8h hi = *(const v8h*)(p + 16);
  v16h f;
#pragma unroll
  for (int i = 0; i < 8; i++) { f[i] = lo[i]; f[i + 8] = hi[i]; }
  return f;
}

// A fragment from an LDS tile with row stride `str`
__device__ __forceinline__ v16h lds_a_frag(const _Float16* base, int row0, int str, int lane) {
  int r = lane & 15;
  int sel = lane >> 4;
  const _Float16* p = base + (row0 + r) * str + sel * 8;
  v8h lo = *(const v8h*)p;
  v8h hi = *(const v8h*)(p + 16);
  v16h f;
#pragma unroll
  for (int i = 0; i < 8; i++) { f[i] = lo[i]; f[i + 8] = hi[i]; }
  return f;
}

// B 32x16 f16 layout: lane<16 -> col=lane, K chunks [k0,k0+8) and [k0+16,k0+24); lane>=16 -> chunks shifted by +8
__device__ __forceinline__ v16h lds_b_frag(const _Float16* base, int col0, int str, int lane) {
  int n = lane & 15;
  int sel = lane >> 4;
  const _Float16* p = base + (col0 + n) * str + sel * 8;
  v8h lo = *(const v8h*)p;
  v8h hi = *(const v8h*)(p + 16);
  v16h f;
#pragma unroll
  for (int i = 0; i < 8; i++) { f[i] = lo[i]; f[i + 8] = hi[i]; }
  return f;
}

// ---------------- GEMM: Y[M,N] = A[M,K] * W[N,K]^T ----------------
// 128x128 block tile, 8 waves. Double-buffered LDS slabs (A,B: 128x32 f16)
// filled with async global->LDS; WMMA consumes slab k while slab k+1 streams in.
template <bool F16OUT>
__global__ __launch_bounds__(256) void gemm_xwt(const _Float16* __restrict__ A,
                                                const _Float16* __restrict__ W,
                                                void* __restrict__ Y,
                                                int M, int N, int K) {
  __shared__ _Float16 a_lds[2][128 * ASTR];
  __shared__ _Float16 b_lds[2][128 * ASTR];

  const int lane = threadIdx.x & 31;
  const int w    = threadIdx.x >> 5;
  const int rowB = blockIdx.y * 128;
  const int colB = blockIdx.x * 128;
  const int wm   = (w >> 1) * 32;
  const int wn   = (w & 1) * 64;

  auto stage = [&](int buf, int k0) {
#pragma unroll
    for (int i = 0; i < 2; i++) {
      int c  = threadIdx.x + i * 256;     // 512 16B-chunks per 128x32 slab
      int r  = c >> 2;
      int cb = (c & 3) * 8;
      async_ld_b128(A + (size_t)(rowB + r) * K + k0 + cb, &a_lds[buf][r * ASTR + cb]);
      async_ld_b128(W + (size_t)(colB + r) * K + k0 + cb, &b_lds[buf][r * ASTR + cb]);
    }
  };

  v8f acc[2][4];
#pragma unroll
  for (int i = 0; i < 2; i++)
#pragma unroll
    for (int j = 0; j < 4; j++)
#pragma unroll
      for (int e = 0; e < 8; e++) acc[i][j][e] = 0.0f;

  stage(0, 0);
  for (int k0 = 0; k0 < K; k0 += KSTEP) {
    const int buf = (k0 >> 5) & 1;
    wait_async0();
    __syncthreads();                      // slab `buf` ready
    if (k0 + KSTEP < K) stage(buf ^ 1, k0 + KSTEP);

    v16h a0 = lds_a_frag(&a_lds[buf][0], wm,      ASTR, lane);
    v16h a1 = lds_a_frag(&a_lds[buf][0], wm + 16, ASTR, lane);
    v16h b[4];
#pragma unroll
    for (int j = 0; j < 4; j++) b[j] = lds_b_frag(&b_lds[buf][0], wn + j * 16, ASTR, lane);
#pragma unroll
    for (int j = 0; j < 4; j++) {
      acc[0][j] = __builtin_amdgcn_wmma_f32_16x16x32_f16(false, a0, false, b[j],
                                                         (short)0, acc[0][j], false, false);
      acc[1][j] = __builtin_amdgcn_wmma_f32_16x16x32_f16(false, a1, false, b[j],
                                                         (short)0, acc[1][j], false, false);
    }
    __syncthreads();                      // reads done before slab restaged
  }

  const int half = lane >> 4;
  const int n = lane & 15;
#pragma unroll
  for (int i = 0; i < 2; i++)
#pragma unroll
    for (int j = 0; j < 4; j++)
#pragma unroll
      for (int e = 0; e < 8; e++) {
        size_t r = (size_t)(rowB + wm + i * 16 + half * 8 + e);
        size_t c = (size_t)(colB + wn + j * 16 + n);
        float v = acc[i][j][e];
        if (F16OUT) ((_Float16*)Y)[r * N + c] = (_Float16)v;
        else        ((float*)Y)[r * N + c]    = v;
      }
}

// ---------------- Flash attention (causal), one (b,h,64-row q-block) per WG ----
__global__ __launch_bounds__(128) void attn_kernel(const _Float16* __restrict__ Qh,
                                                   const _Float16* __restrict__ Kh,
                                                   const _Float16* __restrict__ Vh,
                                                   _Float16* __restrict__ Ctx) {
  __shared__ _Float16 k_lds[64 * LDSTR];        // K tile, row-major [k][d]
  __shared__ _Float16 vt_lds[64 * LDSTR];       // V tile transposed [d][k]
  __shared__ _Float16 p_lds[4][16 * LDSTR];     // per-wave P transpose buffer

  const int lane = threadIdx.x & 31;
  const int w    = threadIdx.x >> 5;
  const int qb   = blockIdx.x;
  const int bh   = blockIdx.y;
  const int b = bh >> 4, h = bh & 15;
  const size_t base_bs = (size_t)b * SEQ;
  const int headoff = h * DK;
  const int qrow0 = qb * 64 + w * 16;
  const int half = lane >> 4;
  const int n = lane & 15;

  // Q fragments, pre-scaled by 1/sqrt(Dk)=0.125 (exact in f16)
  const _Float16* Qbase = Qh + (base_bs + qrow0) * D_MODEL + headoff;
  v16h qa[2];
#pragma unroll
  for (int t = 0; t < 2; t++) {
    qa[t] = load_a_frag(Qbase, D_MODEL, 0, t * 32, lane);
#pragma unroll
    for (int e = 0; e < 16; e++) qa[t][e] = qa[t][e] * (_Float16)0.125f;
  }

  float m[8], l[8];
  v8f o[4];
#pragma unroll
  for (int e = 0; e < 8; e++) { m[e] = -3.0e38f; l[e] = 0.0f; }
#pragma unroll
  for (int t = 0; t < 4; t++)
#pragma unroll
    for (int e = 0; e < 8; e++) o[t][e] = 0.0f;

  const int nkb = qb + 1;  // causal: same k-block count for all waves in the WG
  for (int kb = 0; kb < nkb; ++kb) {
    __syncthreads();  // previous tile fully consumed before overwrite
    {
      const int krow0 = kb * 64;
      // K tile via async global->LDS
      for (int c = threadIdx.x; c < 512; c += 128) {
        int r = c >> 3, cb = (c & 7) * 8;
        async_ld_b128(Kh + (base_bs + krow0 + r) * D_MODEL + headoff + cb,
                      &k_lds[r * LDSTR + cb]);
      }
      // V tile with transpose-on-store (async DMA cannot transpose)
      for (int c = threadIdx.x; c < 512; c += 128) {
        int r = c >> 3, cb = (c & 7) * 8;
        v8h vv = *(const v8h*)(Vh + (base_bs + krow0 + r) * D_MODEL + headoff + cb);
#pragma unroll
        for (int j = 0; j < 8; j++) vt_lds[(cb + j) * LDSTR + r] = vv[j];
      }
      if (kb + 1 < nkb) {  // hint next causal block toward L2
        __builtin_prefetch((const void*)(Kh + (base_bs + (kb + 1) * 64 + (threadIdx.x >> 1)) * D_MODEL + headoff), 0, 2);
        __builtin_prefetch((const void*)(Vh + (base_bs + (kb + 1) * 64 + (threadIdx.x >> 1)) * D_MODEL + headoff), 0, 2);
      }
      wait_async0();
    }
    __syncthreads();

    // ---- S = (Q*scale) K^T : load all 8 B-frags, then 8 back-to-back WMMAs ----
    v16h bf[4][2];
#pragma unroll
    for (int t = 0; t < 4; t++)
#pragma unroll
      for (int dk = 0; dk < 2; dk++) {
        const _Float16* p = &k_lds[(t * 16 + n) * LDSTR + dk * 32 + half * 8];
        v8h lo = *(const v8h*)p; v8h hi = *(const v8h*)(p + 16);
#pragma unroll
        for (int e = 0; e < 8; e++) { bf[t][dk][e] = lo[e]; bf[t][dk][e + 8] = hi[e]; }
      }
    v8f s[4];
#pragma unroll
    for (int t = 0; t < 4; t++) {
#pragma unroll
      for (int e = 0; e < 8; e++) s[t][e] = 0.0f;
      s[t] = __builtin_amdgcn_wmma_f32_16x16x32_f16(false, qa[0], false, bf[t][0],
                                                    (short)0, s[t], false, false);
      s[t] = __builtin_amdgcn_wmma_f32_16x16x32_f16(false, qa[1], false, bf[t][1],
                                                    (short)0, s[t], false, false);
    }

    if (kb == qb) {  // mask above diagonal
#pragma unroll
      for (int t = 0; t < 4; t++) {
        int col = kb * 64 + t * 16 + n;
#pragma unroll
        for (int e = 0; e < 8; e++) {
          int row = qrow0 + half * 8 + e;
          if (col > row) s[t][e] = -3.0e38f;
        }
      }
    }

    // Online softmax; xor-shuffles {1,2,4,8} reduce within 16-lane halves
#pragma unroll
    for (int e = 0; e < 8; e++) {
      float mx = fmaxf(fmaxf(s[0][e], s[1][e]), fmaxf(s[2][e], s[3][e]));
      mx = fmaxf(mx, __shfl_xor(mx, 1));
      mx = fmaxf(mx, __shfl_xor(mx, 2));
      mx = fmaxf(mx, __shfl_xor(mx, 4));
      mx = fmaxf(mx, __shfl_xor(mx, 8));
      float mn = fmaxf(m[e], mx);
      float alpha = __expf(m[e] - mn);
      m[e] = mn;
      float pv[4], rs = 0.0f;
#pragma unroll
      for (int t = 0; t < 4; t++) { pv[t] = __expf(s[t][e] - mn); rs += pv[t]; }
      rs += __shfl_xor(rs, 1); rs += __shfl_xor(rs, 2);
      rs += __shfl_xor(rs, 4); rs += __shfl_xor(rs, 8);
      l[e] = alpha * l[e] + rs;
#pragma unroll
      for (int t = 0; t < 4; t++) {
        o[t][e] *= alpha;
        p_lds[w][(half * 8 + e) * LDSTR + t * 16 + n] = (_Float16)pv[t];
      }
    }
    __syncthreads();  // P visible

    // ---- O += P*V : load both A-frags + all 8 V B-frags, then 8 WMMAs ----
    v16h pa[2];
#pragma unroll
    for (int kc = 0; kc < 2; kc++) {
      const _Float16* p = &p_lds[w][n * LDSTR + kc * 32 + half * 8];
      v8h lo = *(const v8h*)p; v8h hi = *(const v8h*)(p + 16);
#pragma unroll
      for (int e = 0; e < 8; e++) { pa[kc][e] = lo[e]; pa[kc][e + 8] = hi[e]; }
    }
    v16h vb[4][2];
#pragma unroll
    for (int td = 0; td < 4; ++td)
#pragma unroll
      for (int kc = 0; kc < 2; ++kc) {
        const _Float16* p = &vt_lds[(td * 16 + n) * LDSTR + kc * 32 + half * 8];
        v8h lo = *(const v8h*)p; v8h hi = *(const v8h*)(p + 16);
#pragma unroll
        for (int e = 0; e < 8; e++) { vb[td][kc][e] = lo[e]; vb[td][kc][e + 8] = hi[e]; }
      }
#pragma unroll
    for (int td = 0; td < 4; ++td) {
      o[td] = __builtin_amdgcn_wmma_f32_16x16x32_f16(false, pa[0], false, vb[td][0],
                                                     (short)0, o[td], false, false);
      o[td] = __builtin_amdgcn_wmma_f32_16x16x32_f16(false, pa[1], false, vb[td][1],
                                                     (short)0, o[td], false, false);
    }
  }

  // epilogue: divide by l, store f16 context for the final projection
#pragma unroll
  for (int e = 0; e < 8; e++) {
    float inv = 1.0f / l[e];
#pragma unroll
    for (int t = 0; t < 4; t++) {
      size_t r = base_bs + qrow0 + half * 8 + e;
      Ctx[r * D_MODEL + headoff + t * 16 + n] = (_Float16)(o[t][e] * inv);
    }
  }
}

// ---------------- host-side launch ----------------
extern "C" void kernel_launch(void* const* d_in, const int* in_sizes, int n_in,
                              void* d_out, int out_size, void* d_ws, size_t ws_size,
                              hipStream_t stream) {
  (void)in_sizes; (void)n_in; (void)out_size; (void)ws_size;
  const float* x  = (const float*)d_in[0];
  const float* Wq = (const float*)d_in[1];
  const float* Wk = (const float*)d_in[2];
  const float* Wv = (const float*)d_in[3];
  const float* Wo = (const float*)d_in[4];
  float* out = (float*)d_out;

  char* ws = (char*)d_ws;
  size_t off = 0;
  auto carve = [&](size_t elems) {
    _Float16* p = (_Float16*)(ws + off);
    off += elems * sizeof(_Float16);
    return p;
  };
  const size_t nx = (size_t)M_TOTAL * D_MODEL;       // 8.39M
  const size_t nw = (size_t)D_MODEL * D_MODEL;       // 1.05M
  _Float16* x_h   = carve(nx);
  _Float16* wq_h  = carve(nw);
  _Float16* wk_h  = carve(nw);
  _Float16* wv_h  = carve(nw);
  _Float16* wo_h  = carve(nw);
  _Float16* q_h   = carve(nx);
  _Float16* k_h   = carve(nx);
  _Float16* v_h   = carve(nx);
  _Float16* ctx_h = carve(nx);

  cast_f16_kernel<<<2048, 256, 0, stream>>>(x,  x_h,  (int)nx);
  cast_f16_kernel<<<1024, 256, 0, stream>>>(Wq, wq_h, (int)nw);
  cast_f16_kernel<<<1024, 256, 0, stream>>>(Wk, wk_h, (int)nw);
  cast_f16_kernel<<<1024, 256, 0, stream>>>(Wv, wv_h, (int)nw);
  cast_f16_kernel<<<1024, 256, 0, stream>>>(Wo, wo_h, (int)nw);

  dim3 ggrid(D_MODEL / 128, M_TOTAL / 128);  // (8, 64)
  gemm_xwt<true><<<ggrid, 256, 0, stream>>>(x_h, wq_h, q_h, M_TOTAL, D_MODEL, D_MODEL);
  gemm_xwt<true><<<ggrid, 256, 0, stream>>>(x_h, wk_h, k_h, M_TOTAL, D_MODEL, D_MODEL);
  gemm_xwt<true><<<ggrid, 256, 0, stream>>>(x_h, wv_h, v_h, M_TOTAL, D_MODEL, D_MODEL);

  attn_kernel<<<dim3(SEQ / 64, BATCH * NHEADS), 128, 0, stream>>>(q_h, k_h, v_h, ctx_h);

  gemm_xwt<false><<<ggrid, 256, 0, stream>>>(ctx_h, wo_h, out, M_TOTAL, D_MODEL, D_MODEL);
}
